// TiSASRec_88261577933501
// MI455X (gfx1250) — matrix operand order repaired and checked by reading, not compiled, not run on hardware
//
#include <hip/hip_runtime.h>
#include <hip/hip_bf16.h>

#define B_  8
#define L_  256
#define H_  128
#define NH_ 2
#define D_  64
#define QT_ 16

typedef __attribute__((ext_vector_type(16))) __bf16 v16bf;
typedef __attribute__((ext_vector_type(8)))  __bf16 v8bf;
typedef __attribute__((ext_vector_type(8)))  float  v8f;
typedef __attribute__((ext_vector_type(4)))  float  fx4;

// ---------------------------------------------------------------------------
// Kernel 1: fused projections.
//   Q  = queries @ Wq.T + bq                (f32 + bf16 copies)
//   KP = keys    @ Wk.T + bk + abs_pos_K    (bf16)
//   VP = keys    @ Wv.T + bv + abs_pos_V    (bf16)
// stored per-head: [B][NH][L][D]
// ---------------------------------------------------------------------------
__global__ __launch_bounds__(256) void tisas_proj(
    const float* __restrict__ queries, const float* __restrict__ keys,
    const float* __restrict__ posK, const float* __restrict__ posV,
    const float* __restrict__ Wq, const float* __restrict__ bq,
    const float* __restrict__ Wk, const float* __restrict__ bk,
    const float* __restrict__ Wv, const float* __restrict__ bv,
    float* __restrict__ Qf, __bf16* __restrict__ Qb,
    __bf16* __restrict__ KP, __bf16* __restrict__ VP)
{
  int idx = blockIdx.x * 256 + threadIdx.x;        // [0, B*L*H)
  int col = idx & (H_ - 1);
  int row = idx >> 7;                              // [0, B*L)
  const float* xq = queries + (size_t)row * H_;
  const float* xk = keys    + (size_t)row * H_;
  const float* wq = Wq + (size_t)col * H_;
  const float* wk = Wk + (size_t)col * H_;
  const float* wv = Wv + (size_t)col * H_;
  float q = bq[col], k = bk[col], v = bv[col];
#pragma unroll 8
  for (int i = 0; i < H_; ++i) {
    float a = xq[i], c = xk[i];
    q = fmaf(a, wq[i], q);
    k = fmaf(c, wk[i], k);
    v = fmaf(c, wv[i], v);
  }
  int b    = row >> 8;
  int l    = row & (L_ - 1);
  int head = col >> 6;
  int d    = col & (D_ - 1);
  size_t o = (((size_t)(b * NH_ + head) * L_) + l) * D_ + d;
  Qf[o] = q;
  Qb[o] = (__bf16)q;
  KP[o] = (__bf16)(k + posK[idx]);
  VP[o] = (__bf16)(v + posV[idx]);
}

// ---------------------------------------------------------------------------
// Kernel 2: fused three-term attention for one (b, head, 16-query tile).
// ---------------------------------------------------------------------------
__global__ __launch_bounds__(256) void tisas_attn(
    const float* __restrict__ Qf, const __bf16* __restrict__ Qb,
    const __bf16* __restrict__ KP, const __bf16* __restrict__ VP,
    const float* __restrict__ Tk, const float* __restrict__ Tv,
    const unsigned char* __restrict__ tmask,
    float* __restrict__ out)
{
  __shared__ __align__(16) float  sQ [QT_][D_];   // 4 KB  Q tile f32
  __shared__ __align__(16) __bf16 sQb[QT_][D_];   // 2 KB  Q tile bf16
  __shared__ __align__(16) float  sS [QT_][L_];   // 16 KB logits -> probs f32
  __shared__ __align__(16) __bf16 sP [QT_][L_];   // 8 KB  probs bf16
  __shared__ __align__(16) float  sO [QT_][D_];   // 4 KB  output accum
  __shared__ float red[QT_][16];
  __shared__ float rmax[QT_], rsum[QT_];

  const int tid  = threadIdx.x;
  const int wave = tid >> 5;
  const int lane = tid & 31;
  const int bid  = blockIdx.x;                    // [0, 256)
  const int qt   = bid & 15;
  const int head = (bid >> 4) & (NH_ - 1);
  const int b    = bid >> 5;
  const int q0   = qt * QT_;

  const size_t headbase = (size_t)(b * NH_ + head) * L_ * D_;

  // ---- Phase 0: load Q tile (f32 + bf16), zero output accumulator ----
  {
    const float* qsrc = Qf + headbase + (size_t)q0 * D_;
    fx4 t = *reinterpret_cast<const fx4*>(qsrc + tid * 4);
    reinterpret_cast<fx4*>(&sQ[0][0])[tid] = t;
    __bf16* qb = &sQb[0][0];
    qb[tid * 4 + 0] = (__bf16)t.x;
    qb[tid * 4 + 1] = (__bf16)t.y;
    qb[tid * 4 + 2] = (__bf16)t.z;
    qb[tid * 4 + 3] = (__bf16)t.w;
    fx4 z = {0.f, 0.f, 0.f, 0.f};
    reinterpret_cast<fx4*>(&sO[0][0])[tid] = z;
  }
  __syncthreads();

  // ---- Phase 1a: logits = Q @ (K+Pk)^T via v_wmma_f32_16x16x32_bf16 ----
  // Each wave owns key-tiles {wave, wave+8}; manually unrolled so the WMMAs
  // sit in straight-line code with structurally full EXEC.
  {
    const int row = lane & 15;
    const int hi8 = (lane >> 4) * 16;
#pragma unroll
    for (int it = 0; it < 2; ++it) {
      const int kt = wave + it * 8;
      v8f acc = {};
#pragma unroll
      for (int dc = 0; dc < 2; ++dc) {
        // A: 16x32 bf16, contiguous K (lanes 0-15: K 0-15; 16-31: K 16-31)
        const int kb = dc * 32 + hi8;
        v8bf lo = *reinterpret_cast<const v8bf*>(&sQb[row][kb]);
        v8bf hi = *reinterpret_cast<const v8bf*>(&sQb[row][kb + 8]);
        v16bf a;
#pragma unroll
        for (int i = 0; i < 8; ++i) { a[i] = lo[i]; a[i + 8] = hi[i]; }
        // B: 32x16 bf16, lane = key column; B[kdim, n] = KP[key=n, d=kdim]
        const int key = kt * 16 + (lane & 15);
        const int db  = dc * 32 + (lane >> 4) * 16;
        v16bf bm = *reinterpret_cast<const v16bf*>(KP + headbase + (size_t)key * D_ + db);
        acc = __builtin_amdgcn_wmma_f32_16x16x32_bf16(false, a, false, bm,
                                                      (short)0, acc, false, false);
      }
#pragma unroll
      for (int r = 0; r < 8; ++r)
        sS[r + (lane >> 4) * 8][kt * 16 + (lane & 15)] = acc[r];
    }
  }
  __syncthreads();

  // ---- Phase 1b: logits += <Tk[b,q,k,head,:], Q[q,:]>  (bandwidth-bound) ----
  // thread t owns key k = t for all 16 queries; non-temporal streaming reads,
  // with a prefetch across the 128 KB stride to the next query row.
  {
    const int k = tid;
    const float* base0 = Tk + (((size_t)(b * L_)) * L_ + k) * H_ + head * D_
                            + (size_t)q0 * L_ * H_;
#pragma unroll 1
    for (int j = 0; j < QT_; ++j) {
      const float* base = base0 + (size_t)j * L_ * H_;
      if (j + 1 < QT_)
        __builtin_prefetch(base + (size_t)L_ * H_, 0, 0);  // global_prefetch_b8
      fx4 acc = {0.f, 0.f, 0.f, 0.f};
#pragma unroll
      for (int d4 = 0; d4 < 16; ++d4) {
        fx4 t  = __builtin_nontemporal_load(reinterpret_cast<const fx4*>(base) + d4);
        fx4 qv = reinterpret_cast<const fx4*>(&sQ[j][0])[d4];
        acc.x = fmaf(t.x, qv.x, acc.x);
        acc.y = fmaf(t.y, qv.y, acc.y);
        acc.z = fmaf(t.z, qv.z, acc.z);
        acc.w = fmaf(t.w, qv.w, acc.w);
      }
      sS[j][k] += (acc.x + acc.y) + (acc.z + acc.w);
    }
  }
  __syncthreads();

  // ---- Phase 2: scale 1/sqrt(d), mask (time_mask row | causal), softmax ----
  {
    const int q   = tid >> 4;
    const int l16 = tid & 15;
    const int qglob = q0 + q;
    const bool rowmasked = tmask[b * L_ + qglob] != 0;
    const float NEG = -4294967295.0f;              // -(2^32)+1
    float lmax = -3.0e38f;
    for (int kk = l16; kk < L_; kk += 16) {
      float v = sS[q][kk] * 0.125f;
      if (rowmasked || kk > qglob) v = NEG;
      sS[q][kk] = v;
      lmax = fmaxf(lmax, v);
    }
    red[q][l16] = lmax;
    __syncthreads();
    if (l16 == 0) {
      float m = red[q][0];
#pragma unroll
      for (int i = 1; i < 16; ++i) m = fmaxf(m, red[q][i]);
      rmax[q] = m;
    }
    __syncthreads();
    const float m = rmax[q];
    float lsum = 0.f;
    for (int kk = l16; kk < L_; kk += 16) {
      float e = __expf(sS[q][kk] - m);
      sS[q][kk] = e;
      lsum += e;
    }
    red[q][l16] = lsum;
    __syncthreads();
    if (l16 == 0) {
      float s = 0.f;
#pragma unroll
      for (int i = 0; i < 16; ++i) s += red[q][i];
      rsum[q] = s;
    }
    __syncthreads();
    const float inv = 1.0f / rsum[q];
    for (int kk = l16; kk < L_; kk += 16) {
      float p = sS[q][kk] * inv;
      sS[q][kk] = p;
      sP[q][kk] = (__bf16)p;
    }
  }
  __syncthreads();

  // ---- Phase 3a: out += sum_k p[q,k] * Tv[b,q,k,head,:]  (bandwidth-bound) ----
  // thread owns (q, 4-float chunk of d); 16 lanes cover one 256B d-slice.
  {
    const int q  = tid >> 4;
    const int d4 = tid & 15;
    const float* base = Tv + (((size_t)(b * L_) + (q0 + q)) * L_) * H_ + head * D_ + d4 * 4;
    __builtin_prefetch(base, 0, 0);
    fx4 acc = {0.f, 0.f, 0.f, 0.f};
#pragma unroll 4
    for (int k = 0; k < L_; ++k) {
      float p = sS[q][k];
      fx4 t = __builtin_nontemporal_load(
          reinterpret_cast<const fx4*>(base + (size_t)k * H_));
      acc.x = fmaf(p, t.x, acc.x);
      acc.y = fmaf(p, t.y, acc.y);
      acc.z = fmaf(p, t.z, acc.z);
      acc.w = fmaf(p, t.w, acc.w);
    }
    reinterpret_cast<fx4*>(&sO[q][0])[d4] = acc;   // exclusive owner (sO was 0)
  }
  __syncthreads();

  // ---- Phase 3b: out += attn @ (V+Pv) via WMMA; waves 0-3 own 16-wide n-tiles ----
  if (wave < 4) {
    const int nt  = wave;
    const int row = lane & 15;
    const int hi8 = (lane >> 4) * 16;
    v8f acc = {};
#pragma unroll
    for (int kc = 0; kc < 8; ++kc) {
      const int kb = kc * 32 + hi8;
      v8bf lo = *reinterpret_cast<const v8bf*>(&sP[row][kb]);
      v8bf hi = *reinterpret_cast<const v8bf*>(&sP[row][kb + 8]);
      v16bf a;
#pragma unroll
      for (int i = 0; i < 8; ++i) { a[i] = lo[i]; a[i + 8] = hi[i]; }
      const int dcol = nt * 16 + (lane & 15);
      const int krow = kc * 32 + (lane >> 4) * 16;
      const __bf16* vp = VP + headbase + (size_t)krow * D_ + dcol;
      v16bf bm;
#pragma unroll
      for (int i = 0; i < 16; ++i) bm[i] = vp[(size_t)i * D_];
      acc = __builtin_amdgcn_wmma_f32_16x16x32_bf16(false, a, false, bm,
                                                    (short)0, acc, false, false);
    }
#pragma unroll
    for (int r = 0; r < 8; ++r)
      sO[r + (lane >> 4) * 8][nt * 16 + (lane & 15)] += acc[r];
  }
  __syncthreads();

  // ---- Phase 4: write out[b, l, head*64 + d] ----
  {
    const int q  = tid >> 4;
    const int d4 = tid & 15;
    fx4 v = reinterpret_cast<const fx4*>(&sO[q][0])[d4];
    float* o = out + ((size_t)(b * L_) + (q0 + q)) * H_ + head * D_ + d4 * 4;
    *reinterpret_cast<fx4*>(o) = v;
  }
}

// ---------------------------------------------------------------------------
extern "C" void kernel_launch(void* const* d_in, const int* in_sizes, int n_in,
                              void* d_out, int out_size, void* d_ws, size_t ws_size,
                              hipStream_t stream) {
  (void)in_sizes; (void)n_in; (void)out_size; (void)ws_size;
  const float* queries = (const float*)d_in[0];
  const float* keys    = (const float*)d_in[1];
  const unsigned char* tmask = (const unsigned char*)d_in[2];
  // d_in[3] attn_mask == ~tril (strict upper triangle): applied analytically.
  const float* Tk   = (const float*)d_in[4];
  const float* Tv   = (const float*)d_in[5];
  const float* posK = (const float*)d_in[6];
  const float* posV = (const float*)d_in[7];
  const float* Wq = (const float*)d_in[8];
  const float* bq = (const float*)d_in[9];
  const float* Wk = (const float*)d_in[10];
  const float* bk = (const float*)d_in[11];
  const float* Wv = (const float*)d_in[12];
  const float* bv = (const float*)d_in[13];
  // d_in[14] head_num == 2 (compile-time NH_)
  float* out = (float*)d_out;

  // workspace: Qf (1 MB f32) | Qb | KP | VP (512 KB bf16 each) = 2.5 MB
  char* ws = (char*)d_ws;
  float*  Qf = (float*)ws;
  __bf16* Qb = (__bf16*)(ws + (1u << 20));
  __bf16* KP = (__bf16*)(ws + (1u << 20) + (1u << 19));
  __bf16* VP = (__bf16*)(ws + (1u << 20) + 2u * (1u << 19));

  tisas_proj<<<dim3((B_ * L_ * H_) / 256), 256, 0, stream>>>(
      queries, keys, posK, posV, Wq, bq, Wk, bk, Wv, bv, Qf, Qb, KP, VP);
  tisas_attn<<<dim3(B_ * NH_ * (L_ / QT_)), 256, 0, stream>>>(
      Qf, Qb, KP, VP, Tk, Tv, tmask, out);
}
